// mLSTMCell_33526514712650
// MI455X (gfx1250) — matrix operand order repaired and checked by reading, not compiled, not run on hardware
//
#include <hip/hip_runtime.h>

typedef __bf16 bf16;
typedef __attribute__((ext_vector_type(16))) __bf16 v16bf;
typedef __attribute__((ext_vector_type(8)))  float  v8f;

#define HD 512          // hidden size
#define BM 64           // batch rows per workgroup
#define LSTR 520        // padded LDS row stride in bf16 elements (1040 B, 16B aligned)
#define WG_THREADS 256  // 8 wave32 waves

#define V8F_ZERO ((v8f){0.f, 0.f, 0.f, 0.f, 0.f, 0.f, 0.f, 0.f})

// ---------------- fragment loaders (layouts per CDNA5 ISA) ----------------

// 16-bit A matrix 16x32: lanes 0-15 = rows M, chunk0 K=[hi*8..hi*8+7] in v0-3,
// chunk1 K=[16+hi*8..+7] in v4-7  -> two 16B LDS loads 32B apart.
__device__ __forceinline__ v16bf load_a_lds(const bf16* base, int row0, int kk, int lane) {
  int r  = row0 + (lane & 15);
  int hi = lane >> 4;
  const uint4* p = (const uint4*)(base + (size_t)r * LSTR + kk * 32 + hi * 8);
  union { v16bf v; uint4 q[2]; } u;
  u.q[0] = p[0];   // K = kk*32 + hi*8 .. +7
  u.q[1] = p[2];   // K = kk*32 + 16 + hi*8 .. +7
  return u.v;
}

// 16-bit B matrix 32x16: lane&15 = column N, same K order as A: K=[hi*8..+7] then
// K=[16+hi*8..+7]. Weights stored transposed Wt[n][k] -> two 16B loads 32B apart.
__device__ __forceinline__ v16bf load_b_wt(const bf16* wt, int n0, int kk, int lane) {
  int n  = n0 + (lane & 15);
  int hi = lane >> 4;
  const uint4* p = (const uint4*)(wt + (size_t)n * HD + kk * 32 + hi * 8);
  union { v16bf v; uint4 q[2]; } u;
  u.q[0] = p[0];
  u.q[1] = p[2];
  return u.v;
}

__device__ __forceinline__ v8f wmma_bf16(v16bf a, v16bf b, v8f c) {
  return __builtin_amdgcn_wmma_f32_16x16x32_bf16(false, a, false, b, (short)0, c, false, false);
}

__device__ __forceinline__ float sigm_f(float x) { return 1.0f / (1.0f + __expf(-x)); }
__device__ __forceinline__ float tanh_f(float x) {
  float e = __expf(-2.0f * x);
  return (1.0f - e) / (1.0f + e);
}

// 4 row-tiles x 4 col-tiles, K=512: per k-step 4 A + 4 B fragment loads feed 16 WMMAs.
__device__ __forceinline__ void gemm16(const bf16* A, const bf16* W, int cw, int lane,
                                       v8f* acc) {
  for (int kk = 0; kk < 16; ++kk) {
    v16bf a[4];
#pragma unroll
    for (int r = 0; r < 4; ++r) a[r] = load_a_lds(A, r * 16, kk, lane);
    v16bf b[4];
#pragma unroll
    for (int c = 0; c < 4; ++c) b[c] = load_b_wt(W, cw + c * 16, kk, lane);
#pragma unroll
    for (int r = 0; r < 4; ++r)
#pragma unroll
      for (int c = 0; c < 4; ++c)
        acc[r * 4 + c] = wmma_bf16(a[r], b[c], acc[r * 4 + c]);
  }
}

// Two gates x 2 col-tiles x 4 row-tiles: per k-step 8 A (CSE'd when A0==A1) + 4 B loads
// feed 16 WMMAs.
__device__ __forceinline__ void gemm8x2(const bf16* A0, const bf16* W0,
                                        const bf16* A1, const bf16* W1,
                                        int n0, int lane, v8f* acc0, v8f* acc1) {
  for (int kk = 0; kk < 16; ++kk) {
    v16bf a0[4], a1[4];
#pragma unroll
    for (int r = 0; r < 4; ++r) {
      a0[r] = load_a_lds(A0, r * 16, kk, lane);
      a1[r] = load_a_lds(A1, r * 16, kk, lane);
    }
    v16bf b0[2], b1[2];
#pragma unroll
    for (int c = 0; c < 2; ++c) {
      b0[c] = load_b_wt(W0, n0 + c * 16, kk, lane);
      b1[c] = load_b_wt(W1, n0 + c * 16, kk, lane);
    }
#pragma unroll
    for (int r = 0; r < 4; ++r)
#pragma unroll
      for (int c = 0; c < 2; ++c) {
        acc0[r * 2 + c] = wmma_bf16(a0[r], b0[c], acc0[r * 2 + c]);
        acc1[r * 2 + c] = wmma_bf16(a1[r], b1[c], acc1[r * 2 + c]);
      }
  }
}

// ---------------- weight prep: fp32 W[k][n] -> bf16 Wt[n][k], 10 matrices ----------------

__global__ void prep_weights(const float* __restrict__ w0, const float* __restrict__ w1,
                             const float* __restrict__ w2, const float* __restrict__ w3,
                             const float* __restrict__ w4, const float* __restrict__ w5,
                             const float* __restrict__ w6, const float* __restrict__ w7,
                             const float* __restrict__ w8, const float* __restrict__ w9,
                             bf16* __restrict__ outw) {
  const float* ws[10] = {w0, w1, w2, w3, w4, w5, w6, w7, w8, w9};
  const float* in = ws[blockIdx.z];
  bf16* o = outw + (size_t)blockIdx.z * HD * HD;

  __shared__ float tile[32][33];
  int tx = threadIdx.x, ty = threadIdx.y;
  int bx = blockIdx.x * 32, by = blockIdx.y * 32;
#pragma unroll
  for (int i = 0; i < 32; i += 8)
    tile[ty + i][tx] = in[(size_t)(by + ty + i) * HD + (bx + tx)];
  __syncthreads();
#pragma unroll
  for (int i = 0; i < 32; i += 8) {
    int n = bx + ty + i;
    int k = by + tx;
    o[(size_t)n * HD + k] = (bf16)tile[tx][ty + i];
  }
}

// ---------------- fused mLSTM cell ----------------

__global__ __launch_bounds__(WG_THREADS)
void mlstm_kernel(const float* __restrict__ x, const float* __restrict__ hprev,
                  const float* __restrict__ cprev, const bf16* __restrict__ wts,
                  const float* __restrict__ bm, const float* __restrict__ bi,
                  const float* __restrict__ bff, const float* __restrict__ bo,
                  const float* __restrict__ bc, float* __restrict__ out) {
  extern __shared__ __attribute__((aligned(16))) bf16 smem[];
  bf16* xs  = smem;                  // 64 x 520 : x tile (bf16)
  bf16* hsm = smem + BM * LSTR;      // 64 x 520 : h tile
  bf16* xts = smem + 2 * BM * LSTR;  // 64 x 520 : x_tilde tile
  bf16* cs  = smem + 3 * BM * LSTR;  // 64 x 520 : running cell state (per-wave stripes)

  const int tid     = threadIdx.x;
  const int lane    = tid & 31;
  const int w       = tid >> 5;      // wave id 0..7
  const int cw      = w * 64;        // unique 64-column stripe per wave (no B redundancy)
  const int hi      = lane >> 4;
  const int colLane = lane & 15;
  const int wgrow   = blockIdx.x * BM;

  const bf16* Wm = wts + 0 * (size_t)HD * HD;
  const bf16* Um = wts + 1 * (size_t)HD * HD;
  const bf16* Wi = wts + 2 * (size_t)HD * HD;
  const bf16* Ui = wts + 3 * (size_t)HD * HD;
  const bf16* Wf = wts + 4 * (size_t)HD * HD;
  const bf16* Uf = wts + 5 * (size_t)HD * HD;
  const bf16* Wo = wts + 6 * (size_t)HD * HD;
  const bf16* Uo = wts + 7 * (size_t)HD * HD;
  const bf16* Wc = wts + 8 * (size_t)HD * HD;
  const bf16* Uc = wts + 9 * (size_t)HD * HD;

  // ---- stage x,h (fp32 -> bf16) into LDS: b128 global reads, packed b64 LDS writes ----
  {
    const float4* xg = (const float4*)(x + (size_t)wgrow * HD);
    const float4* hg = (const float4*)(hprev + (size_t)wgrow * HD);
    for (int idx = tid; idx < BM * HD / 4; idx += WG_THREADS) {
      int row = idx >> 7;            // HD/4 = 128 float4 per row
      int c4  = (idx & 127) << 2;
      float4 v = xg[idx];
      union { bf16 e[4]; unsigned long long u; } pk;
      pk.e[0] = (bf16)v.x; pk.e[1] = (bf16)v.y; pk.e[2] = (bf16)v.z; pk.e[3] = (bf16)v.w;
      *(unsigned long long*)(xs + (size_t)row * LSTR + c4) = pk.u;
      v = hg[idx];
      pk.e[0] = (bf16)v.x; pk.e[1] = (bf16)v.y; pk.e[2] = (bf16)v.z; pk.e[3] = (bf16)v.w;
      *(unsigned long long*)(hsm + (size_t)row * LSTR + c4) = pk.u;
    }
  }
  __syncthreads();

  // ---- pass 1: m = x@Wm + h@Um + bm ; x_tilde = x * m  (wave: 64 rows x 64 cols) ----
  {
    v8f accm[16];
#pragma unroll
    for (int t = 0; t < 16; ++t) accm[t] = V8F_ZERO;
    gemm16(xs,  Wm, cw, lane, accm);
    gemm16(hsm, Um, cw, lane, accm);
#pragma unroll
    for (int c = 0; c < 4; ++c) {
      int col = cw + c * 16 + colLane;
      float bmv = bm[col];
#pragma unroll
      for (int r = 0; r < 4; ++r)
#pragma unroll
        for (int v = 0; v < 8; ++v) {    // C layout: element (v + 8*hi, lane&15)
          int row = r * 16 + v + 8 * hi;
          float mval = accm[r * 4 + c][v] + bmv;
          float xv = (float)xs[(size_t)row * LSTR + col];
          xts[(size_t)row * LSTR + col] = (bf16)(xv * mval);
        }
    }
  }
  __syncthreads();   // x_tilde is read across the full K range by all waves

  // ---- gate f: cs = sigmoid(x@Wf + h@Uf + bf) * c_prev ----
  {
    v8f accf[16];
#pragma unroll
    for (int t = 0; t < 16; ++t) accf[t] = V8F_ZERO;
    gemm16(xs,  Wf, cw, lane, accf);
    gemm16(hsm, Uf, cw, lane, accf);
#pragma unroll
    for (int c = 0; c < 4; ++c) {
      int col = cw + c * 16 + colLane;
      float bfv = bff[col];
#pragma unroll
      for (int r = 0; r < 4; ++r)
#pragma unroll
        for (int v = 0; v < 8; ++v) {
          int rowl = r * 16 + v + 8 * hi;
          float cpv = cprev[(size_t)(wgrow + rowl) * HD + col];
          cs[(size_t)rowl * LSTR + col] =
              (bf16)(sigm_f(accf[r * 4 + c][v] + bfv) * cpv);
        }
    }
  }

  // ---- gates i & c: cs += sigmoid(x_tilde@Wi + h@Ui + bi) * tanh(x@Wc + h@Uc + bc) ----
  // (cs stripes are per-wave private: no barrier needed between gate phases)
  for (int cb = 0; cb < 4; cb += 2) {
    v8f acci[8], accc[8];
#pragma unroll
    for (int t = 0; t < 8; ++t) { acci[t] = V8F_ZERO; accc[t] = V8F_ZERO; }
    gemm8x2(xts, Wi, xs,  Wc, cw + cb * 16, lane, acci, accc);
    gemm8x2(hsm, Ui, hsm, Uc, cw + cb * 16, lane, acci, accc);
#pragma unroll
    for (int cc = 0; cc < 2; ++cc) {
      int col = cw + (cb + cc) * 16 + colLane;
      float biv = bi[col], bcv = bc[col];
#pragma unroll
      for (int r = 0; r < 4; ++r)
#pragma unroll
        for (int v = 0; v < 8; ++v) {
          int rowl = r * 16 + v + 8 * hi;
          float s = (float)cs[(size_t)rowl * LSTR + col];
          s += sigm_f(acci[r * 2 + cc][v] + biv) * tanh_f(accc[r * 2 + cc][v] + bcv);
          cs[(size_t)rowl * LSTR + col] = (bf16)s;
        }
    }
  }

  // ---- gate o: out = sigmoid(x@Wo + h@Uo + bo) * tanh(c_t) ----
  {
    v8f acco[16];
#pragma unroll
    for (int t = 0; t < 16; ++t) acco[t] = V8F_ZERO;
    gemm16(xs,  Wo, cw, lane, acco);
    gemm16(hsm, Uo, cw, lane, acco);
#pragma unroll
    for (int c = 0; c < 4; ++c) {
      int col = cw + c * 16 + colLane;
      float bov = bo[col];
#pragma unroll
      for (int r = 0; r < 4; ++r)
#pragma unroll
        for (int v = 0; v < 8; ++v) {
          int rowl = r * 16 + v + 8 * hi;
          float ct = (float)cs[(size_t)rowl * LSTR + col];
          out[(size_t)(wgrow + rowl) * HD + col] =
              sigm_f(acco[r * 4 + c][v] + bov) * tanh_f(ct);
        }
    }
  }
}

// ---------------- launch ----------------

extern "C" void kernel_launch(void* const* d_in, const int* in_sizes, int n_in,
                              void* d_out, int out_size, void* d_ws, size_t ws_size,
                              hipStream_t stream) {
  const float* x = (const float*)d_in[0];
  const float* h = (const float*)d_in[1];
  const float* c = (const float*)d_in[2];
  bf16* wts = (bf16*)d_ws;   // 10 * 512 * 512 bf16 = 5 MB (L2-resident)

  prep_weights<<<dim3(HD / 32, HD / 32, 10), dim3(32, 8, 1), 0, stream>>>(
      (const float*)d_in[3], (const float*)d_in[4], (const float*)d_in[5],
      (const float*)d_in[6], (const float*)d_in[7], (const float*)d_in[8],
      (const float*)d_in[9], (const float*)d_in[10], (const float*)d_in[11],
      (const float*)d_in[12], wts);

  int Bsz = in_sizes[0] / HD;                                  // 32768
  size_t lds_bytes = (size_t)4 * BM * LSTR * sizeof(bf16);     // 260 KB of 320 KB WGP LDS
  mlstm_kernel<<<dim3(Bsz / BM), dim3(WG_THREADS), lds_bytes, stream>>>(
      x, h, c, wts, (const float*)d_in[13], (const float*)d_in[14],
      (const float*)d_in[15], (const float*)d_in[16], (const float*)d_in[17],
      (float*)d_out);
}
